// HyperbolicKinematicAttention_14104672600227
// MI455X (gfx1250) — matrix operand order repaired and checked by reading, not compiled, not run on hardware
//
#include <hip/hip_runtime.h>
#include <math.h>

typedef __attribute__((ext_vector_type(16))) _Float16 v16h;
typedef __attribute__((ext_vector_type(8)))  _Float16 v8h;
typedef __attribute__((ext_vector_type(8)))  float    v8f;

static constexpr int   Bb   = 8;
static constexpr int   Nn   = 2048;
static constexpr int   Dd   = 256;
static constexpr int   Mm   = Bb * Nn;          // 16384 total rows
static constexpr float FEPS = 1e-6f;

// ---------------------------------------------------------------------------
// WMMA fragment loaders (layouts per CDNA5 ISA 7.12.2, wave32)
// A 16x32 f16: lane L -> M=L&15, hi=L>>4 ; slot j -> K = (j>>3)*16 + hi*8 + (j&7)
//   => two contiguous 8-half chunks at [kb+hi*8] and [kb+16+hi*8]
// B 32x16 f16 (from row-major src[n][k], B[k][n]=src[n][k]): lane L -> N=L&15,
//   hi=L>>4 ; slot j -> K = kb + (j>>3)*16 + hi*8 + (j&7) (same as A)
// C/D f32 16x16: lane L -> N=L&15 ; vgpr r -> M = r + 8*(L>>4)
// ---------------------------------------------------------------------------
__device__ __forceinline__ v16h load_a_h(const _Float16* base, int ld, int Mi, int hi, int kb) {
  const _Float16* p0 = base + (size_t)Mi * ld + kb + hi * 8;
  v8h c0 = *(const v8h*)p0;
  v8h c1 = *(const v8h*)(p0 + 16);
  v16h a;
#pragma unroll
  for (int j = 0; j < 8; ++j) { a[j] = c0[j]; a[8 + j] = c1[j]; }
  return a;
}

__device__ __forceinline__ v16h load_a_f(const float* base, int ld, int Mi, int hi, int kb) {
  const float* p0 = base + (size_t)Mi * ld + kb + hi * 8;
  v8f c0 = *(const v8f*)p0;
  v8f c1 = *(const v8f*)(p0 + 16);
  v16h a;
#pragma unroll
  for (int j = 0; j < 8; ++j) { a[j] = (_Float16)c0[j]; a[8 + j] = (_Float16)c1[j]; }
  return a;
}

__device__ __forceinline__ v16h load_b_h(const _Float16* base, int ld, int Ni, int hi, int kb) {
  const _Float16* p = base + (size_t)Ni * ld + kb + hi * 8;
  v8h c0 = *(const v8h*)p;
  v8h c1 = *(const v8h*)(p + 16);
  v16h b;
#pragma unroll
  for (int j = 0; j < 8; ++j) { b[j] = c0[j]; b[8 + j] = c1[j]; }
  return b;
}

__device__ __forceinline__ v16h load_b_f(const float* base, int ld, int Ni, int hi, int kb) {
  const float* p = base + (size_t)Ni * ld + kb + hi * 8;
  v8f c0 = *(const v8f*)p;
  v8f c1 = *(const v8f*)(p + 16);
  v16h b;
#pragma unroll
  for (int j = 0; j < 8; ++j) { b[j] = (_Float16)c0[j]; b[8 + j] = (_Float16)c1[j]; }
  return b;
}

__device__ __forceinline__ v8f wmma_f16(v16h a, v16h b, v8f c) {
  return __builtin_amdgcn_wmma_f32_16x16x32_f16(false, a, false, b, (short)0, c, false, false);
}

// Block-wide sum over 256 threads (8 wave32)
__device__ __forceinline__ float block_sum(float v, float* red) {
  int lane = threadIdx.x & 31, wid = threadIdx.x >> 5;
#pragma unroll
  for (int o = 16; o > 0; o >>= 1) v += __shfl_xor(v, o, 32);
  if (lane == 0) red[wid] = v;
  __syncthreads();
  float s = 0.f;
#pragma unroll
  for (int i = 0; i < 8; ++i) s += red[i];
  return s;
}

__device__ __forceinline__ float acosh_pos(float x) {  // x >= 1+eps
  return logf(x + sqrtf(x * x - 1.0f));
}

// ---------------------------------------------------------------------------
// K1: log-map0 + transport-to-origin. One block per row.
// ---------------------------------------------------------------------------
__global__ __launch_bounds__(256) void k_prep(const float* __restrict__ x,
                                              const float* __restrict__ xv,
                                              float* __restrict__ xtan_out,
                                              _Float16* __restrict__ xt16,
                                              _Float16* __restrict__ vel16) {
  __shared__ float red[8];
  int R = blockIdx.x, t = threadIdx.x;
  const float* xr = x  + (size_t)R * 257;
  const float* vr = xv + (size_t)R * 257;
  float x0 = xr[0], v0 = vr[0];
  float sp = xr[1 + t];
  float s  = block_sum(sp * sp, red);
  float nrm = sqrtf(fmaxf(s, FEPS));
  float xc  = fmaxf(x0, 1.0f + FEPS);
  float dl  = acosh_pos(xc);
  float xt  = (dl / nrm) * sp;
  size_t o = (size_t)R * Dd + t;
  xtan_out[o] = xt;
  xt16[o]     = (_Float16)xt;
  float coef = -v0 / (1.0f + x0);
  vel16[o] = (_Float16)(vr[1 + t] + coef * sp);
}

// ---------------------------------------------------------------------------
// K2: one projection  O = A @ W^T + b  via WMMA f16, fp32 accumulate.
// grid (M/16, 4), block 128 (4 waves, each one 16x16 tile). Launched 5x with
// direct global pointers (no pointer arrays -> loads stay global_*, not flat).
// ---------------------------------------------------------------------------
__global__ __launch_bounds__(128) void k_proj(const _Float16* __restrict__ A,
                                              const float* __restrict__ W,
                                              const float* __restrict__ bias,
                                              _Float16* __restrict__ O) {
  int R  = blockIdx.x * 16;
  int w  = threadIdx.x >> 5;
  int cb = (blockIdx.y * 4 + w) * 16;
  int lane = threadIdx.x & 31, Li = lane & 15, hi = lane >> 4;

  const float* Wc = W + (size_t)cb * Dd;
  v8f acc{};
#pragma unroll
  for (int kb = 0; kb < Dd; kb += 32) {
    v16h a = load_a_h(A + (size_t)R * Dd, Dd, Li, hi, kb);
    v16h b = load_b_f(Wc, Dd, Li, hi, kb);
    acc = wmma_f16(a, b, acc);
  }
  float bia = bias[cb + Li];
#pragma unroll
  for (int r = 0; r < 8; ++r) {
    int row = R + r + 8 * hi;
    O[(size_t)row * Dd + cb + Li] = (_Float16)(acc[r] + bia);
  }
}

// ---------------------------------------------------------------------------
// K3a: exp-map scaling: out0[R]=cosh||row||, row *= sinh||row||/||row||.
// K3b: row sum-of-squares only.
// ---------------------------------------------------------------------------
__global__ __launch_bounds__(256) void k_expscale(_Float16* __restrict__ buf,
                                                  float* __restrict__ out0) {
  __shared__ float red[8];
  int R = blockIdx.x, t = threadIdx.x;
  size_t o = (size_t)R * Dd + t;
  float v = (float)buf[o];
  float s = block_sum(v * v, red);
  float nrm = sqrtf(fmaxf(s, FEPS));
  if (t == 0) out0[R] = coshf(nrm);
  buf[o] = (_Float16)((sinhf(nrm) / nrm) * v);
}

__global__ __launch_bounds__(256) void k_sumsq(const _Float16* __restrict__ buf,
                                               float* __restrict__ out0) {
  __shared__ float red[8];
  int R = blockIdx.x, t = threadIdx.x;
  float v = (float)buf[(size_t)R * Dd + t];
  float s = block_sum(v * v, red);
  if (t == 0) out0[R] = s;
}

// ---------------------------------------------------------------------------
// K4: V[b][m][d] -> V^T[b][d][m] (f16), 32x32 LDS tiles.
// ---------------------------------------------------------------------------
__global__ void k_transpose(const _Float16* __restrict__ vs16, _Float16* __restrict__ vt16) {
  __shared__ _Float16 tile[32][33];
  int b = blockIdx.z;
  int m0 = blockIdx.x * 32, d0 = blockIdx.y * 32;
  int tx = threadIdx.x, ty = threadIdx.y;
#pragma unroll
  for (int i = 0; i < 4; ++i)
    tile[ty + i * 8][tx] = vs16[((size_t)b * Nn + m0 + ty + i * 8) * Dd + d0 + tx];
  __syncthreads();
#pragma unroll
  for (int i = 0; i < 4; ++i)
    vt16[((size_t)b * Dd + d0 + ty + i * 8) * Nn + m0 + tx] = tile[tx][ty + i * 8];
}

// ---------------------------------------------------------------------------
// K5: logits + softmax + attn. Block: 16 q-rows x all 2048 keys.
// 8 waves, wave w owns columns [w*256, w*256+256). Dyn LDS: 16x2048 f32 logits.
// ---------------------------------------------------------------------------
__global__ __launch_bounds__(256) void k_attn(const _Float16* __restrict__ qsp, const _Float16* __restrict__ qvl,
                                              const _Float16* __restrict__ ksp, const _Float16* __restrict__ kvl,
                                              const float* __restrict__ q0g, const float* __restrict__ k0g,
                                              const float* __restrict__ qsqg, const float* __restrict__ ksqg,
                                              const float* __restrict__ tau, float* __restrict__ attn_out) {
  extern __shared__ float smem[];
  float* logits = smem;                 // 16*2048
  float* q0s    = smem + 16 * Nn;       // 16
  float* qsqs   = q0s + 16;             // 16

  int R = blockIdx.x * 16;
  int b = blockIdx.x >> 7;              // /128
  int w = threadIdx.x >> 5;
  int lane = threadIdx.x & 31, Li = lane & 15, hi = lane >> 4;

  if (threadIdx.x < 16) {
    q0s[threadIdx.x]  = q0g[R + threadIdx.x];
    qsqs[threadIdx.x] = qsqg[R + threadIdx.x];
  }
  __syncthreads();

  float inv_tau = 1.0f / fmaxf(tau[0], 1e-3f);

  // Preload the 16x256 A fragments (Q-space and Q-velocity) for this block.
  v16h aq[8], av[8];
#pragma unroll
  for (int kf = 0; kf < 8; ++kf) {
    aq[kf] = load_a_h(qsp + (size_t)R * Dd, Dd, Li, hi, kf * 32);
    av[kf] = load_a_h(qvl + (size_t)R * Dd, Dd, Li, hi, kf * 32);
  }
  const _Float16* kspb = ksp + (size_t)b * Nn * Dd;
  const _Float16* kvlb = kvl + (size_t)b * Nn * Dd;

  for (int t = 0; t < 16; ++t) {
    int cb = w * 256 + t * 16;          // key column base (within batch)
    // prefetch next tile's key rows (global_prefetch_b8)
    if (t < 15) {
      __builtin_prefetch(kspb + (size_t)(cb + 16 + Li) * Dd, 0, 3);
      __builtin_prefetch(kvlb + (size_t)(cb + 16 + Li) * Dd, 0, 3);
    }
    v8f accS{}, accT{};
#pragma unroll
    for (int kf = 0; kf < 8; ++kf) {
      v16h bS = load_b_h(kspb + (size_t)cb * Dd, Dd, Li, hi, kf * 32);
      v16h bT = load_b_h(kvlb + (size_t)cb * Dd, Dd, Li, hi, kf * 32);
      accS = wmma_f16(aq[kf], bS, accS);
      accT = wmma_f16(av[kf], bT, accT);
    }
    float k0v  = k0g [(size_t)b * Nn + cb + Li];
    float ksqv = ksqg[(size_t)b * Nn + cb + Li];
#pragma unroll
    for (int r = 0; r < 8; ++r) {
      int r8 = r + 8 * hi;
      float negI = q0s[r8] * k0v - accS[r];            // -(Minkowski inner)
      float arg  = fmaxf(negI, 1.0f + FEPS);
      float dL   = acosh_pos(arg);
      float pen  = fmaxf(qsqs[r8] + ksqv - 2.0f * accT[r], 0.0f);
      logits[r8 * Nn + cb + Li] = -inv_tau * dL * dL - pen;
    }
  }
  __syncthreads();

  // Softmax: wave w handles rows 2w, 2w+1.
#pragma unroll
  for (int rr = 2 * w; rr <= 2 * w + 1; ++rr) {
    float mx = -3.4e38f;
    for (int c = lane; c < Nn; c += 32) mx = fmaxf(mx, logits[rr * Nn + c]);
#pragma unroll
    for (int o = 16; o > 0; o >>= 1) mx = fmaxf(mx, __shfl_xor(mx, o, 32));
    float sum = 0.f;
    for (int c = lane; c < Nn; c += 32) sum += __expf(logits[rr * Nn + c] - mx);
#pragma unroll
    for (int o = 16; o > 0; o >>= 1) sum += __shfl_xor(sum, o, 32);
    float rs = 1.0f / sum;
    float* dst = attn_out + (size_t)(R + rr) * Nn;
    for (int c = lane; c < Nn; c += 32) dst[c] = __expf(logits[rr * Nn + c] - mx) * rs;
  }
}

// ---------------------------------------------------------------------------
// K6: agg = attn @ V  (K=2048).  attn fp32 -> f16 on load, V^T B-frags.
// Block 256 = 8 waves; wave w owns columns [w*32, w*32+32) (two 16-col tiles).
// ---------------------------------------------------------------------------
__global__ __launch_bounds__(256) void k_av(const float* __restrict__ attn_out,
                                            const _Float16* __restrict__ vt16,
                                            float* __restrict__ agg) {
  int R = blockIdx.x * 16;
  int b = blockIdx.x >> 7;
  int w = threadIdx.x >> 5;
  int lane = threadIdx.x & 31, Li = lane & 15, hi = lane >> 4;
  int cb = w * 32;

  const float*    Ab = attn_out + (size_t)R * Nn;
  const _Float16* Vb = vt16 + (size_t)b * Dd * Nn;

  v8f acc0{}, acc1{};
  for (int kb = 0; kb < Nn; kb += 32) {
    if (kb + 32 < Nn)  // prefetch next attn K-block for this lane's row
      __builtin_prefetch(Ab + (size_t)Li * Nn + kb + 32, 0, 3);
    v16h a  = load_a_f(Ab, Nn, Li, hi, kb);
    v16h b0 = load_b_h(Vb + (size_t)cb * Nn, Nn, Li, hi, kb);
    v16h b1 = load_b_h(Vb + (size_t)(cb + 16) * Nn, Nn, Li, hi, kb);
    acc0 = wmma_f16(a, b0, acc0);
    acc1 = wmma_f16(a, b1, acc1);
  }
#pragma unroll
  for (int r = 0; r < 8; ++r) {
    int row = R + r + 8 * hi;
    agg[(size_t)row * Dd + cb + Li]      = acc0[r];
    agg[(size_t)row * Dd + cb + 16 + Li] = acc1[r];
  }
}

// ---------------------------------------------------------------------------
// K7: z = exp_map0(agg). One block per row; z has 257 components.
// ---------------------------------------------------------------------------
__global__ __launch_bounds__(256) void k_zmap(const float* __restrict__ agg, float* __restrict__ z_out) {
  __shared__ float red[8];
  int R = blockIdx.x, t = threadIdx.x;
  float a = agg[(size_t)R * Dd + t];
  float s = block_sum(a * a, red);
  float nrm = sqrtf(fmaxf(s, FEPS));
  float* zr = z_out + (size_t)R * 257;
  if (t == 0) zr[0] = coshf(nrm);
  zr[1 + t] = (sinhf(nrm) / nrm) * a;
}

// ---------------------------------------------------------------------------
extern "C" void kernel_launch(void* const* d_in, const int* in_sizes, int n_in,
                              void* d_out, int out_size, void* d_ws, size_t ws_size,
                              hipStream_t stream) {
  const float* x    = (const float*)d_in[0];
  const float* x_vel= (const float*)d_in[1];
  const float* Wq   = (const float*)d_in[2];
  const float* bq   = (const float*)d_in[3];
  const float* Wk   = (const float*)d_in[4];
  const float* bk   = (const float*)d_in[5];
  const float* Wv   = (const float*)d_in[6];
  const float* bv   = (const float*)d_in[7];
  const float* tau  = (const float*)d_in[8];

  float* z_out    = (float*)d_out;
  float* attn_out = z_out + (size_t)Mm * 257;
  float* xtan_out = attn_out + (size_t)Mm * Nn;

  char* w = (char*)d_ws;
  const size_t H = (size_t)Mm * Dd * sizeof(_Float16);   // 8 MB per f16 buffer
  _Float16* xt16  = (_Float16*)(w + 0 * H);
  _Float16* vel16 = (_Float16*)(w + 1 * H);
  _Float16* qs16  = (_Float16*)(w + 2 * H);
  _Float16* ks16  = (_Float16*)(w + 3 * H);
  _Float16* vs16  = (_Float16*)(w + 4 * H);
  _Float16* qv16  = (_Float16*)(w + 5 * H);
  _Float16* kv16  = (_Float16*)(w + 6 * H);
  _Float16* vt16  = (_Float16*)(w + 7 * H);
  float* q0  = (float*)(w + 8 * H);
  float* k0  = q0  + Mm;
  float* qsq = k0  + Mm;
  float* ksq = qsq + Mm;
  float* agg = ksq + Mm;                                  // 16 MB fp32

  k_prep<<<Mm, 256, 0, stream>>>(x, x_vel, xtan_out, xt16, vel16);

  dim3 gproj(Mm / 16, 4);
  k_proj<<<gproj, 128, 0, stream>>>(xt16,  Wq, bq, qs16);
  k_proj<<<gproj, 128, 0, stream>>>(xt16,  Wk, bk, ks16);
  k_proj<<<gproj, 128, 0, stream>>>(xt16,  Wv, bv, vs16);
  k_proj<<<gproj, 128, 0, stream>>>(vel16, Wq, bq, qv16);
  k_proj<<<gproj, 128, 0, stream>>>(vel16, Wk, bk, kv16);

  k_expscale<<<Mm, 256, 0, stream>>>(qs16, q0);
  k_expscale<<<Mm, 256, 0, stream>>>(ks16, k0);
  k_sumsq  <<<Mm, 256, 0, stream>>>(qv16, qsq);
  k_sumsq  <<<Mm, 256, 0, stream>>>(kv16, ksq);

  k_transpose<<<dim3(Nn / 32, Dd / 32, Bb), dim3(32, 8), 0, stream>>>(vs16, vt16);

  size_t shmem = (size_t)(16 * Nn + 32) * sizeof(float);  // 128 KB + stats (WGP LDS = 320 KB)
  k_attn<<<Mm / 16, 256, shmem, stream>>>(qs16, qv16, ks16, kv16, q0, k0, qsq, ksq, tau, attn_out);
  k_av<<<Mm / 16, 256, 0, stream>>>(attn_out, vt16, agg);
  k_zmap<<<Mm, 256, 0, stream>>>(agg, z_out);
}
